// CrossAttentionComplexQ_28166395528048
// MI455X (gfx1250) — matrix operand order repaired and checked by reading, not compiled, not run on hardware
//
#include <hip/hip_runtime.h>
#include <hip/hip_bf16.h>

typedef __bf16 bf16_t;
typedef __attribute__((ext_vector_type(16))) __bf16 v16bf;
typedef __attribute__((ext_vector_type(8)))  __bf16 v8bf;
typedef __attribute__((ext_vector_type(8)))  float  v8f;
typedef __attribute__((ext_vector_type(4)))  float  v4f;

#define BQ   64
#define SQ   8
#define SKV  4096
#define DQ   256
#define DKV  512
#define MT   64             // kv rows per workgroup tile (2 WGs co-resident/WGP)
#define PKV  (DKV + 8)      // 520  LDS pitch for kv tile (bf16 elems)
#define PK   (DQ + 8)       // 264  LDS pitch for k_tile (kvrow-major)
#define PV   (MT + 8)       // 72   LDS pitch for v_tile^T (dq-major)
#define PQ   (DQ + 8)       // 264  LDS pitch for q
#define PS   (MT + 8)       // 72   LDS pitch for scores

// ---------------- WMMA helpers -----------------------------------------

static __device__ inline v8f wmma_bf16(v16bf a, v16bf b, v8f c) {
  // D = A(16x32) * B(32x16) + C, fp32 accumulate
  return __builtin_amdgcn_wmma_f32_16x16x32_bf16(
      /*neg_a=*/false, a, /*neg_b=*/false, b,
      /*c_mod=*/(short)0, c, /*reuse_a=*/false, /*reuse_b=*/false);
}

// A-operand (16x32 bf16): lane<16 holds row (lane), K = kb+{0..7, 16..23};
// lane>=16 holds row (lane-16), K = kb+{8..15, 24..31}.
static __device__ inline v16bf ldA(const bf16_t* base, int pitch, int kb, int lane) {
  int row = lane & 15;
  int off = row * pitch + kb + ((lane >> 4) << 3);
  v8bf lo = *(const v8bf*)(base + off);
  v8bf hi = *(const v8bf*)(base + off + 16);
  return __builtin_shufflevector(lo, hi, 0,1,2,3,4,5,6,7,8,9,10,11,12,13,14,15);
}

// B-operand (32x16 bf16): lane<16 holds column (lane), K = kb+{0..7, 16..23};
// lane>=16 holds column (lane-16), K = kb+{8..15, 24..31}.
static __device__ inline v16bf ldB(const bf16_t* base, int pitch, int kb, int lane) {
  int col = lane & 15;
  int off = col * pitch + kb + ((lane >> 4) << 3);
  v8bf lo = *(const v8bf*)(base + off);
  v8bf hi = *(const v8bf*)(base + off + 16);
  return __builtin_shufflevector(lo, hi, 0,1,2,3,4,5,6,7,8,9,10,11,12,13,14,15);
}

// ---------------- Prolog 1: rotate + LayerNorm q -> bf16 ----------------

__global__ __launch_bounds__(256) void qprep_kernel(
    const float* __restrict__ q, const float* __restrict__ fc,
    const float* __restrict__ fs, const float* __restrict__ lw,
    const float* __restrict__ lb, bf16_t* __restrict__ qout) {
  int row = blockIdx.x;           // b*8 + s, 512 rows total
  int s   = row & 7;
  int t   = threadIdx.x;          // 0..255 = feature index
  int p   = t >> 1;               // complex pair index
  float a  = q[row * DQ + 2 * p];
  float bb = q[row * DQ + 2 * p + 1];
  float c  = fc[s * (DQ / 2) + p];
  float sn = fs[s * (DQ / 2) + p];
  float x  = (t & 1) ? (a * sn + bb * c) : (a * c - bb * sn);

  // LayerNorm reduction: wave32 shuffle + cross-wave via LDS
  float sum = x, sq = x * x;
  #pragma unroll
  for (int o = 1; o < 32; o <<= 1) {
    sum += __shfl_xor(sum, o);
    sq  += __shfl_xor(sq, o);
  }
  __shared__ float red[16];
  int wv = t >> 5, ln = t & 31;
  if (ln == 0) { red[wv] = sum; red[8 + wv] = sq; }
  __syncthreads();
  float ts = 0.f, tq = 0.f;
  #pragma unroll
  for (int i = 0; i < 8; ++i) { ts += red[i]; tq += red[8 + i]; }
  float mu  = ts * (1.0f / DQ);
  float var = tq * (1.0f / DQ) - mu * mu;
  float r   = rsqrtf(var + 1e-5f);
  float y   = (x - mu) * r * lw[t] + lb[t];
  qout[row * DQ + t] = (bf16_t)y;
}

// ---------------- Prolog 2: Wk/Wv fp32 -> bf16 --------------------------

__global__ __launch_bounds__(256) void wconv_kernel(
    const float* __restrict__ Wk, const float* __restrict__ Wv,
    bf16_t* __restrict__ wkb, bf16_t* __restrict__ wvb) {
  int i = blockIdx.x * 256 + threadIdx.x;   // 0 .. 131071
  wkb[i] = (bf16_t)Wk[i];
  wvb[i] = (bf16_t)Wv[i];
}

// ---------------- Main fused kernel -------------------------------------

__global__ __launch_bounds__(256) void attn_main_kernel(
    const float* __restrict__ kv, const bf16_t* __restrict__ qn,
    const bf16_t* __restrict__ wkb, const bf16_t* __restrict__ wvb,
    float* __restrict__ out) {
  __shared__ bf16_t kv_s[MT * PKV];   // kv tile, bf16, row-major     (65 KB)
  __shared__ bf16_t kt_s[MT * PK];    // k_tile, kvrow-major          (33 KB)
  __shared__ bf16_t vt_s[DQ * PV];    // v_tile^T, dq-major           (36 KB)
  __shared__ bf16_t q_s [16 * PQ];    // q, rows 8..15 zero-padded    ( 8 KB)
  __shared__ bf16_t sc_s[16 * PS];    // scores / 16, bf16            ( 2 KB)

  const int tid  = threadIdx.x;
  const int lane = tid & 31;
  const int wv   = tid >> 5;                 // wave id 0..7
  const int b    = blockIdx.x >> 6;          // batch  (64 tiles per batch)
  const int tile = blockIdx.x & 63;          // kv-row tile
  const int r0   = tile * MT;

  // ---- stage 1: kv tile fp32 -> bf16 LDS; q -> LDS (zero-padded) ----
  const float* kvp = kv + ((size_t)b * SKV + r0) * DKV;
  #pragma unroll 4
  for (int i = 0; i < 32; ++i) {
    int idx4 = tid + i * 256;                // 0..8191 float4s
    int row  = idx4 >> 7;                    // 512/4 = 128 float4 per row
    int c4   = (idx4 & 127) << 2;
    v4f d = *(const v4f*)(kvp + row * DKV + c4);
    bf16_t* dst = kv_s + row * PKV + c4;
    dst[0] = (bf16_t)d.x; dst[1] = (bf16_t)d.y;
    dst[2] = (bf16_t)d.z; dst[3] = (bf16_t)d.w;
  }
  {
    int row = tid >> 5;                      // 0..7
    int c   = (tid & 31) * 8;
    v8bf qv = *(const v8bf*)(qn + ((size_t)(b * SQ + row)) * DQ + c);
    *(v8bf*)(q_s + row * PQ + c) = qv;
    v8bf z = {};
    *(v8bf*)(q_s + (row + 8) * PQ + c) = z;  // zero-pad rows 8..15
  }
  __syncthreads();

  // ---- stage 2: K and V projections (bf16 WMMA, fp32 acc) ----
  // wave w owns output columns [32w, 32w+32)
  const int mrow_off = (lane >> 4) << 3;     // 0 or 8 (C-layout row offset)
  #pragma unroll
  for (int nt = 0; nt < 2; ++nt) {
    const int n0 = wv * 32 + nt * 16;
    v8f accK[4], accV[4];
    #pragma unroll
    for (int m = 0; m < 4; ++m) { v8f z = {}; accK[m] = z; accV[m] = z; }

    for (int ks = 0; ks < 16; ++ks) {
      const int kb = ks * 32;
      v16bf bk = ldB(wkb + n0 * DKV, DKV, kb, lane);
      v16bf bv = ldB(wvb + n0 * DKV, DKV, kb, lane);
      #pragma unroll
      for (int m = 0; m < 4; ++m) {
        v16bf a = ldA(kv_s + (m * 16) * PKV, PKV, kb, lane);
        accK[m] = wmma_bf16(a, bk, accK[m]);
        accV[m] = wmma_bf16(a, bv, accV[m]);
      }
    }
    const int ncol = n0 + (lane & 15);
    #pragma unroll
    for (int m = 0; m < 4; ++m) {
      // k_tile: kvrow-major, strided b16 stores (consumer needs rows)
      #pragma unroll
      for (int e = 0; e < 8; ++e) {
        int mr = m * 16 + e + mrow_off;      // kv row within tile
        kt_s[mr * PK + ncol] = (bf16_t)accK[m][e];
      }
      // v_tile^T: 8 consecutive kv-rows land contiguously -> one b128 store
      v8bf pkv_;
      #pragma unroll
      for (int e = 0; e < 8; ++e) pkv_[e] = (bf16_t)accV[m][e];
      *(v8bf*)(vt_s + ncol * PV + m * 16 + mrow_off) = pkv_;
    }
  }
  __syncthreads();

  // ---- stage 3a: scores = q @ k_tile^T, scaled 1/16, -> bf16 LDS ----
  if (wv < 4) {                              // 4 waves cover 64 kv rows
    v8f acc = {};
    const int nrow0 = wv * 16;               // kv-row slice of this wave
    #pragma unroll
    for (int ks = 0; ks < 8; ++ks) {
      v16bf a  = ldA(q_s, PQ, ks * 32, lane);
      v16bf bm = ldB(kt_s + nrow0 * PK, PK, ks * 32, lane);
      acc = wmma_bf16(a, bm, acc);
    }
    const int ncol = nrow0 + (lane & 15);
    #pragma unroll
    for (int e = 0; e < 8; ++e)
      sc_s[(e + mrow_off) * PS + ncol] = (bf16_t)(acc[e] * 0.0625f);
  }
  __syncthreads();

  // ---- stage 3b: out += scores @ v_tile ; atomic accumulate to HBM ----
  #pragma unroll
  for (int nt = 0; nt < 2; ++nt) {
    const int n0 = wv * 32 + nt * 16;        // dq output columns
    v8f acc = {};
    #pragma unroll
    for (int ks = 0; ks < 2; ++ks) {
      v16bf a  = ldA(sc_s, PS, ks * 32, lane);
      v16bf bm = ldB(vt_s + n0 * PV, PV, ks * 32, lane);
      acc = wmma_bf16(a, bm, acc);
    }
    if (lane < 16) {                         // rows 0..7 are real q rows
      float* op = out + (size_t)b * SQ * DQ + (n0 + lane);
      #pragma unroll
      for (int e = 0; e < 8; ++e)
        atomicAdd(op + e * DQ, acc[e]);
    }
  }
}

// ---------------- launch ------------------------------------------------

extern "C" void kernel_launch(void* const* d_in, const int* in_sizes, int n_in,
                              void* d_out, int out_size, void* d_ws, size_t ws_size,
                              hipStream_t stream) {
  const float* q_real = (const float*)d_in[0];
  const float* kv     = (const float*)d_in[1];
  const float* fc     = (const float*)d_in[2];
  const float* fs     = (const float*)d_in[3];
  const float* lw     = (const float*)d_in[4];
  const float* lb     = (const float*)d_in[5];
  const float* Wk     = (const float*)d_in[6];
  const float* Wv     = (const float*)d_in[7];
  float* out = (float*)d_out;

  bf16_t* qn  = (bf16_t*)d_ws;                 // 64*8*256 bf16
  bf16_t* wkb = qn  + (size_t)BQ * SQ * DQ;    // 256*512 bf16
  bf16_t* wvb = wkb + (size_t)DQ * DKV;

  qprep_kernel<<<BQ * SQ, 256, 0, stream>>>(q_real, fc, fs, lw, lb, qn);
  wconv_kernel<<<(DQ * DKV) / 256, 256, 0, stream>>>(Wk, Wv, wkb, wvb);
  hipMemsetAsync(d_out, 0, (size_t)out_size * sizeof(float), stream);
  attn_main_kernel<<<BQ * (SKV / MT), 256, 0, stream>>>(kv, qn, wkb, wvb, out);
}
